// OutlierSmoothing_15522011807856
// MI455X (gfx1250) — matrix operand order repaired and checked by reading, not compiled, not run on hardware
//
#include <hip/hip_runtime.h>

typedef __attribute__((ext_vector_type(16))) _Float16 v16h;
typedef __attribute__((ext_vector_type(4)))  _Float16 v4h;
typedef __attribute__((ext_vector_type(8)))  float    v8f;
typedef __attribute__((ext_vector_type(4)))  float    v4f;

#define N_TOKENS   32768
#define D_MODEL    1024
#define ROW_CHUNKS 64
#define CHUNK      (N_TOKENS / ROW_CHUNKS)   // 512 rows per partial-max block

// Workspace layout (bytes):
//   [0,       262144)  partial column maxima of x: 64 x 1024 f32
//   [262144,  266240)  s      : 1024 f32
//   [266240,  270336)  inv_s  : 1024 f32
//   [270336,  +2 MiB)  wh     : 1024 x 1024 f16 (= s[o] * weight[o][d])
#define WS_PART 0
#define WS_S    262144
#define WS_INV  266240
#define WS_WH   270336

// ---- Pass 1: per-column partial max of x (coalesced, split over row chunks) ----
__global__ void colmax_partial(const float* __restrict__ x, float* __restrict__ part) {
  const int col = blockIdx.x * blockDim.x + threadIdx.x;     // 0..1023
  const size_t r0 = (size_t)blockIdx.y * CHUNK;
  const float* p = x + r0 * D_MODEL + col;
  float m = -__builtin_inff();
  for (int r = 0; r < CHUNK; ++r) m = fmaxf(m, p[(size_t)r * D_MODEL]);
  part[blockIdx.y * D_MODEL + col] = m;
}

// ---- Pass 2: finish x-max, weight column max, s = sqrt(max_act)/sqrt(max_w) ----
__global__ void finalize_scales(const float* __restrict__ part,
                                const float* __restrict__ w,
                                float* __restrict__ s_out,
                                float* __restrict__ inv_out) {
  const int col = blockIdx.x * blockDim.x + threadIdx.x;     // 0..1023
  float ma = -__builtin_inff();
  for (int p = 0; p < ROW_CHUNKS; ++p) ma = fmaxf(ma, part[p * D_MODEL + col]);
  float mw = -__builtin_inff();
  for (int r = 0; r < D_MODEL; ++r) mw = fmaxf(mw, w[(size_t)r * D_MODEL + col]);
  const float s = sqrtf(ma) / sqrtf(mw);                     // ALPHA = 0.5
  s_out[col]   = s;
  inv_out[col] = 1.0f / s;
}

// ---- Pass 3: wh[o][d] = f16(s[o] * weight[o][d])  (2 MiB, stays in L2) ----
__global__ void convert_w(const float* __restrict__ w,
                          const float* __restrict__ s_buf,
                          _Float16* __restrict__ wh) {
  const int row = blockIdx.x;                                // 0..1023
  const float sr = s_buf[row];
  const int d = threadIdx.x * 4;                             // 256 thr * 4 = 1024
  v4f wv = *(const v4f*)(w + (size_t)row * D_MODEL + d);
  v4h hv;
#pragma unroll
  for (int t = 0; t < 4; ++t) hv[t] = (_Float16)(sr * wv[t]);
  *(v4h*)(wh + (size_t)row * D_MODEL + d) = hv;
}

// ---- Pass 4: WMMA GEMM  out = f16(x * inv_s) @ wh^T + bias ----
// Block: 256 threads = 8 waves as 2(M) x 4(N); wave tile 64x64; block tile 128x256.
__global__ void __launch_bounds__(256) gemm_wmma(const float* __restrict__ x,
                                                 const float* __restrict__ inv_s,
                                                 const _Float16* __restrict__ wh,
                                                 const float* __restrict__ bias,
                                                 float* __restrict__ out) {
  const int lane  = threadIdx.x & 31;
  const int wv    = threadIdx.x >> 5;
  const int l15   = lane & 15;
  const int lh    = lane >> 4;          // half-wave select
  const int waveM = wv >> 2;            // 0..1
  const int waveN = wv & 3;             // 0..3
  const int m0 = blockIdx.y * 128 + waveM * 64;   // token row base
  const int o0 = blockIdx.x * 256 + waveN * 64;   // out-channel base

  v8f c[4][4] = {};                     // 16 accumulator fragments (f32)

  for (int k0 = 0; k0 < D_MODEL; k0 += 32) {
    // SmoothQuant scale vectors for this wave's A lanes (A layout: K = 16*lh + 0..15)
    const int da = k0 + 16 * lh;
    const v4f i0 = *(const v4f*)(inv_s + da);
    const v4f i1 = *(const v4f*)(inv_s + da + 4);
    const v4f i2 = *(const v4f*)(inv_s + da + 8);
    const v4f i3 = *(const v4f*)(inv_s + da + 12);

    // B fragments: lane holds col N=l15, halves = K = 16*lh + 0..15 (contiguous 32B of wh row)
    v16h b[4];
#pragma unroll
    for (int ni = 0; ni < 4; ++ni) {
      const int orow = o0 + ni * 16 + l15;
      b[ni] = *(const v16h*)(wh + (size_t)orow * D_MODEL + k0 + 16 * lh);
    }

    // A fragments: read f32 x, rescale by inv_s, convert to f16 in-register
    v16h a[4];
#pragma unroll
    for (int mi = 0; mi < 4; ++mi) {
      const float* xr = x + (size_t)(m0 + mi * 16 + l15) * D_MODEL + da;
      const v4f x0 = *(const v4f*)(xr);
      const v4f x1 = *(const v4f*)(xr + 4);
      const v4f x2 = *(const v4f*)(xr + 8);
      const v4f x3 = *(const v4f*)(xr + 12);
      v16h av;
#pragma unroll
      for (int t = 0; t < 4; ++t) {
        av[t]      = (_Float16)(x0[t] * i0[t]);
        av[4 + t]  = (_Float16)(x1[t] * i1[t]);
        av[8 + t]  = (_Float16)(x2[t] * i2[t]);
        av[12 + t] = (_Float16)(x3[t] * i3[t]);
      }
      a[mi] = av;
    }

#pragma unroll
    for (int mi = 0; mi < 4; ++mi)
#pragma unroll
      for (int ni = 0; ni < 4; ++ni)
        c[mi][ni] = __builtin_amdgcn_wmma_f32_16x16x32_f16(
            /*neg_a=*/false, a[mi], /*neg_b=*/false, b[ni],
            /*c_mod=*/(short)0, c[mi][ni], /*reuse_a=*/false, /*reuse_b=*/false);
  }

  // C/D layout: lane l, vgpr j -> (M = j + 8*lh, N = l15); add bias and store
#pragma unroll
  for (int ni = 0; ni < 4; ++ni) {
    const int o  = o0 + ni * 16 + l15;
    const float bv = bias[o];
#pragma unroll
    for (int mi = 0; mi < 4; ++mi) {
      const int mrow = m0 + mi * 16 + 8 * lh;
#pragma unroll
      for (int j = 0; j < 8; ++j)
        out[(size_t)(mrow + j) * D_MODEL + o] = c[mi][ni][j] + bv;
    }
  }
}

extern "C" void kernel_launch(void* const* d_in, const int* in_sizes, int n_in,
                              void* d_out, int out_size, void* d_ws, size_t ws_size,
                              hipStream_t stream) {
  const float* x    = (const float*)d_in[0];   // [32768,1024]
  const float* w    = (const float*)d_in[1];   // [1024,1024]
  const float* bias = (const float*)d_in[2];   // [1024]
  float* out = (float*)d_out;

  char* ws = (char*)d_ws;
  float*    part    = (float*)(ws + WS_PART);
  float*    s_buf   = (float*)(ws + WS_S);
  float*    inv_buf = (float*)(ws + WS_INV);
  _Float16* wh      = (_Float16*)(ws + WS_WH);

  colmax_partial <<<dim3(D_MODEL / 256, ROW_CHUNKS), 256, 0, stream>>>(x, part);
  finalize_scales<<<D_MODEL / 256, 256, 0, stream>>>(part, w, s_buf, inv_buf);
  convert_w      <<<D_MODEL, 256, 0, stream>>>(w, s_buf, wh);
  gemm_wmma      <<<dim3(D_MODEL / 256, N_TOKENS / 128), 256, 0, stream>>>(
      x, inv_buf, wh, bias, out);
}
